// AutoregressiveFlowLayer_28484223108052
// MI455X (gfx1250) — matrix operand order repaired and checked by reading, not compiled, not run on hardware
//
#include <hip/hip_runtime.h>

// MADE autoregressive flow layer for MI455X (gfx1250), wave32 + WMMA f16 + TDM.
// prep kernel: mask+cvt+transpose weights once per r into d_ws (f16, LDS image).
// main kernel: grid = (B/128, R); 256 threads = 8 waves; wave 0 DMAs the whole
//              62.5KB weight image into LDS with one TENSOR_LOAD_TO_LDS while
//              all waves gather inputs; then 56 WMMAs per wave.

#define D_IN   1024
#define RG     32
#define RMAXC  32
#define H1C    128
#define H2C    128
#define OUTC   64
#define BTILE  128

#define LDX  40    // 32 + 8 halves pad (16B multiple)
#define LDW1 40
#define LDH  136   // 128 + 8 halves pad
#define LDW2 136
#define LDWO 136

// packed weight image (halves): W1t [128*40] | W2t [128*136] | Wot [64*136]
#define W1T_OFF   0
#define W2T_OFF   5120
#define WOT_OFF   22528
#define WIMG_H    31232            // halves  (= 62464 bytes, multiple of 8)
#define WIMG_Q    7808             // 8-byte elements

typedef _Float16 h8_t  __attribute__((ext_vector_type(8)));
typedef _Float16 v16h  __attribute__((ext_vector_type(16)));
typedef float    v8f   __attribute__((ext_vector_type(8)));
typedef unsigned int u32x4 __attribute__((ext_vector_type(4)));
typedef int          i32x8 __attribute__((ext_vector_type(8)));
typedef int          i32x4 __attribute__((ext_vector_type(4)));

// ---------------------------------------------------------------- prep kernel
__global__ __launch_bounds__(256)
void prep_weights(const float* __restrict__ W1,
                  const float* __restrict__ W2,
                  const float* __restrict__ Wout,
                  const unsigned char* __restrict__ M1,
                  const unsigned char* __restrict__ M2,
                  const unsigned char* __restrict__ Mout,
                  _Float16* __restrict__ ws) {
  const int r   = blockIdx.x;
  const int tid = threadIdx.x;
  _Float16* o = ws + (size_t)r * WIMG_H;

  const float* w1 = W1 + r * (RMAXC * H1C);
  const unsigned char* m1 = M1 + r * (RMAXC * H1C);
  for (int i = tid; i < RMAXC * H1C; i += 256) {
    const int d = i >> 7, h = i & 127;
    o[W1T_OFF + h * LDW1 + d] = (_Float16)(m1[i] ? w1[i] : 0.0f);
  }
  const float* w2 = W2 + r * (H1C * H2C);
  const unsigned char* m2 = M2 + r * (H1C * H2C);
  for (int i = tid; i < H1C * H2C; i += 256) {
    const int h = i >> 7, k = i & 127;
    o[W2T_OFF + k * LDW2 + h] = (_Float16)(m2[i] ? w2[i] : 0.0f);
  }
  const float* wo = Wout + r * (H2C * OUTC);
  const unsigned char* mo = Mout + r * (H2C * OUTC);
  for (int i = tid; i < H2C * OUTC; i += 256) {
    const int k = i >> 6, q = i & 63;
    o[WOT_OFF + q * LDWO + k] = (_Float16)(mo[i] ? wo[i] : 0.0f);
  }
}

// ------------------------------------------------------------- WMMA fragments
// A-fragment (16x32 f16, M x K): lane l -> row M = l&15;
// element e -> K = kbase + (e<8?0:16) + (l>=16?8:0) + e%8
__device__ __forceinline__ v16h ld_afrag(const _Float16* t, int row, int ld,
                                         int kbase, int lane) {
  const _Float16* p = t + row * ld + kbase + ((lane & 16) ? 8 : 0);
  h8_t lo = *(const h8_t*)(p);
  h8_t hi = *(const h8_t*)(p + 16);
  v16h a;
#pragma unroll
  for (int e = 0; e < 8; ++e) { a[e] = lo[e]; a[e + 8] = hi[e]; }
  return a;
}

// B-fragment (32x16 f16, K x N) from an N-major (transposed) tile:
// lane l -> col N = l&15; element e -> K = kbase + (e<8?0:16) + (l>=16?8:0) + e%8
__device__ __forceinline__ v16h ld_bfrag(const _Float16* t, int nrow, int ld,
                                         int kbase, int lane) {
  const _Float16* p = t + nrow * ld + kbase + ((lane & 16) ? 8 : 0);
  h8_t lo = *(const h8_t*)(p);
  h8_t hi = *(const h8_t*)(p + 16);
  v16h b;
#pragma unroll
  for (int e = 0; e < 8; ++e) { b[e] = lo[e]; b[e + 8] = hi[e]; }
  return b;
}

// C/D f32 layout: lane l -> N = l&15, VGPR j -> M = j + (l>=16 ? 8 : 0)
__device__ __forceinline__ void st_relu_f16(_Float16* t, int ld, int m0,
                                            int nbase, int lane, v8f acc) {
  const int n  = nbase + (lane & 15);
  const int mb = m0 + ((lane & 16) ? 8 : 0);
#pragma unroll
  for (int j = 0; j < 8; ++j) {
    float v = acc[j];
    v = v > 0.f ? v : 0.f;
    t[(mb + j) * ld + n] = (_Float16)v;
  }
}

// ----------------------------------------------------------------- main kernel
__global__ __launch_bounds__(256)
void made_flow_kernel(const float* __restrict__ inputs,
                      const int* __restrict__ idx,
                      const unsigned char* __restrict__ valid,
                      const _Float16* __restrict__ wimg,
                      float* __restrict__ out) {
  __shared__ __align__(16) _Float16 sWeights[WIMG_H];   // filled by TDM
  __shared__ __align__(16) _Float16 sXh [BTILE * LDX];  // gathered inputs f16
  __shared__ __align__(16) _Float16 sH1 [BTILE * LDH];
  __shared__ __align__(16) _Float16 sH2 [BTILE * LDH];
  __shared__ int   sIdx[RMAXC];
  __shared__ float sVal[RMAXC];

  const _Float16* sW1t = sWeights + W1T_OFF;
  const _Float16* sW2t = sWeights + W2T_OFF;
  const _Float16* sWot = sWeights + WOT_OFF;

  const int tid   = threadIdx.x;
  const int r     = blockIdx.y;
  const int bbase = blockIdx.x * BTILE;

  if (tid < RMAXC) {  // wave 0 only
    sIdx[tid] = idx[r * RMAXC + tid];
    sVal[tid] = valid[r * RMAXC + tid] ? 1.0f : 0.0f;

    // --- one TDM op: copy this r's 62464B weight image from ws into LDS.
    // 1-D tensor: data_size = 8B, tensor_dim0 = tile_dim0 = 7808 elements.
    const unsigned long long ga =
        (unsigned long long)(uintptr_t)(wimg + (size_t)r * WIMG_H);
    const unsigned int lds = (unsigned int)(uintptr_t)(&sWeights[0]);

    u32x4 g0;
    g0[0] = 1u;                                    // count=1, user mode
    g0[1] = lds;                                   // lds_addr
    g0[2] = (unsigned int)(ga & 0xFFFFFFFFu);      // global_addr[31:0]
    g0[3] = (unsigned int)((ga >> 32) & 0x1FFFFFFu) | (2u << 30);  // +type=2

    i32x8 g1;
    g1[0] = 0x00030000;                            // mask=0, data_size=8B
    g1[1] = (int)((WIMG_Q & 0xFFFF) << 16);        // tensor_dim0 lo16
    g1[2] = (int)((WIMG_Q >> 16) & 0xFFFF) | (1 << 16);  // dim0 hi | dim1=1 lo
    g1[3] = (int)(WIMG_Q << 16);                   // dim1 hi(0) | tile_dim0
    g1[4] = 1;                                     // tile_dim1=1, tile_dim2=0
    g1[5] = (int)WIMG_Q;                           // tensor_dim0_stride lo
    g1[6] = 0;
    g1[7] = 0;
    const i32x4 gz4 = {0, 0, 0, 0};
    const i32x8 gz8 = {0, 0, 0, 0, 0, 0, 0, 0};
    __builtin_amdgcn_tensor_load_to_lds(g0, g1, gz4, gz4, gz8, 0);
  }
  __syncthreads();   // publish sIdx/sVal; TDM still in flight

  // ---- gather xg tile (128 rows x 32 features), overlapped with the TDM
#pragma unroll 4
  for (int i = tid; i < BTILE * RMAXC; i += 256) {
    const int row = i >> 5, k = i & 31;
    const float x = inputs[(size_t)(bbase + row) * D_IN + sIdx[k]] * sVal[k];
    sXh[row * LDX + k] = (_Float16)x;
  }
  if (tid < 32) __builtin_amdgcn_s_wait_tensorcnt(0);
  __syncthreads();   // weights + gathered inputs now visible

  const int wave = tid >> 5;
  const int lane = tid & 31;
  const int m0   = wave * 16;

  // ---- layer 1: (16x32) x (32x128)
  {
    const v16h a = ld_afrag(sXh, m0 + (lane & 15), LDX, 0, lane);
#pragma unroll
    for (int n = 0; n < 8; ++n) {
      const v16h b = ld_bfrag(sW1t, n * 16 + (lane & 15), LDW1, 0, lane);
      v8f c = {};
      c = __builtin_amdgcn_wmma_f32_16x16x32_f16(false, a, false, b,
                                                 (short)0, c, false, false);
      st_relu_f16(sH1, LDH, m0, n * 16, lane, c);
    }
  }
  __syncthreads();

  // ---- layer 2: (16x128) x (128x128)
  {
    v16h a[4];
#pragma unroll
    for (int kc = 0; kc < 4; ++kc)
      a[kc] = ld_afrag(sH1, m0 + (lane & 15), LDH, kc * 32, lane);
#pragma unroll
    for (int n = 0; n < 8; ++n) {
      v8f c = {};
#pragma unroll
      for (int kc = 0; kc < 4; ++kc) {
        const v16h b = ld_bfrag(sW2t, n * 16 + (lane & 15), LDW2, kc * 32, lane);
        c = __builtin_amdgcn_wmma_f32_16x16x32_f16(false, a[kc], false, b,
                                                   (short)0, c, false, false);
      }
      st_relu_f16(sH2, LDH, m0, n * 16, lane, c);
    }
  }
  __syncthreads();

  // ---- layer 3: (16x128) x (128x64) + log-likelihood, all in registers
  {
    v16h a[4];
#pragma unroll
    for (int kc = 0; kc < 4; ++kc)
      a[kc] = ld_afrag(sH2, m0 + (lane & 15), LDH, kc * 32, lane);

    float acc_ll[8];
#pragma unroll
    for (int j = 0; j < 8; ++j) acc_ll[j] = 0.f;

    const int n_in = lane & 15;
    const int grp  = (lane & 16) ? 1 : 0;

#pragma unroll
    for (int t = 0; t < 4; ++t) {
      v8f c = {};
#pragma unroll
      for (int kc = 0; kc < 4; ++kc) {
        const v16h b = ld_bfrag(sWot, t * 16 + (lane & 15), LDWO, kc * 32, lane);
        c = __builtin_amdgcn_wmma_f32_16x16x32_f16(false, a[kc], false, b,
                                                   (short)0, c, false, false);
      }
      // out col o = t*16 + n_in: even o -> shift, odd o -> log_s, k = o/2.
      const int   k  = (t * 16 + n_in) >> 1;
      const float vk = sVal[k];
#pragma unroll
      for (int j = 0; j < 8; ++j) {
        const float mine    = c[j];
        const float partner = __shfl_xor(mine, 1, 32);
        const float shiftv  = mine;     // valid in even lanes
        const float logs    = partner;  // valid in even lanes
        const float xg = (float)sXh[(m0 + grp * 8 + j) * LDX + k];
        const float u  = (xg - shiftv) * __expf(-logs);
        acc_ll[j] += (-0.5f * u * u - 0.91893853320467274f - logs) * vk;
      }
    }
#pragma unroll
    for (int j = 0; j < 8; ++j) {
      float s = acc_ll[j];
      s += __shfl_xor(s, 2, 32);
      s += __shfl_xor(s, 4, 32);
      s += __shfl_xor(s, 8, 32);
      acc_ll[j] = s;
    }
    if ((lane & 15) == 0) {  // lane 0 -> rows m0..m0+7, lane 16 -> m0+8..m0+15
#pragma unroll
      for (int j = 0; j < 8; ++j) {
        const int row = bbase + m0 + grp * 8 + j;
        out[(size_t)row * RG + r] = acc_ll[j];
      }
    }
  }
}

extern "C" void kernel_launch(void* const* d_in, const int* in_sizes, int n_in,
                              void* d_out, int out_size, void* d_ws, size_t ws_size,
                              hipStream_t stream) {
  (void)in_sizes; (void)n_in; (void)ws_size; (void)out_size;
  const float* inputs = (const float*)d_in[0];
  const float* W1     = (const float*)d_in[1];
  const float* W2     = (const float*)d_in[2];
  const float* Wout   = (const float*)d_in[3];
  const int*   idx    = (const int*)d_in[4];
  const unsigned char* valid = (const unsigned char*)d_in[5];
  const unsigned char* M1    = (const unsigned char*)d_in[6];
  const unsigned char* M2    = (const unsigned char*)d_in[7];
  const unsigned char* Mout  = (const unsigned char*)d_in[8];
  float* out = (float*)d_out;
  _Float16* wimg = (_Float16*)d_ws;   // needs 32 * 62464 B = ~1.91 MB

  prep_weights<<<dim3(RG), 256, 0, stream>>>(W1, W2, Wout, M1, M2, Mout, wimg);
  made_flow_kernel<<<dim3(8192 / BTILE, RG), 256, 0, stream>>>(
      inputs, idx, valid, wimg, out);
}
